// FusedCombineBmm_14851996909966
// MI455X (gfx1250) — matrix operand order repaired and checked by reading, not compiled, not run on hardware
//
#include <hip/hip_runtime.h>

// ---------------------------------------------------------------------------
// Fused grouped GEMM (MoE experts): out[e] = silu(x[e] @ w[e] + b[e])
//   E=8, M=4096, K=2048, N=8192, fp32 in/out.
// Strategy (MI455X / gfx1250):
//   1) one-pass cast  x  -> xh [E][M][K]  f16           (workspace)
//   2) one-pass cast+transpose w -> wt [E][N][K] f16    (workspace, K-major)
//   3) WMMA GEMM: TDM (tensor_load_to_lds) double-buffered K-major tiles,
//      v_wmma_f32_16x16x32_f16, fused bias+silu epilogue.
// Workspace requirement: E*M*K*2 + E*N*K*2 = 402,653,184 bytes.
// ---------------------------------------------------------------------------

#define E_ 8
#define M_ 4096
#define K_ 2048
#define N_ 8192

#define BM 128          // block tile M
#define BN 256          // block tile N
#define BK 32           // K per WMMA step
#define LDK 40          // padded LDS row pitch in halves (80 B): conflict-free b128 reads
#define NKT (K_ / BK)   // 64 K-steps

typedef __attribute__((ext_vector_type(16))) _Float16 v16h;
typedef __attribute__((ext_vector_type(8)))  _Float16 v8h;
typedef __attribute__((ext_vector_type(4)))  _Float16 v4h;
typedef __attribute__((ext_vector_type(8)))  float    v8f;
typedef __attribute__((ext_vector_type(4)))  float    f32x4;
typedef __attribute__((ext_vector_type(4)))  unsigned u32x4;
typedef __attribute__((ext_vector_type(8)))  int      i32x8;
typedef __attribute__((ext_vector_type(4)))  int      i32x4;

// Zero-cost compiler fence: forbids caching/hoisting LDS loads across it.
#define LDS_COMPILER_FENCE() asm volatile("" ::: "memory")

// --------------------------- TDM tile load ---------------------------------
// 2-D tile: tile_dim0 = BK halves (64 B, contiguous K), tile_dim1 = rows.
// pad_enable with interval=16 dwords (64 B) and amount=4 dwords (16 B) gives
// an LDS row pitch of 80 B == LDK halves.
// `lds_dst` is the C-visible alias of the destination; escaping it through the
// asm makes the TDM's LDS write visible to alias analysis (otherwise the
// compiler folds all smem loads to undef -- observed in round 2 disasm).
__device__ __forceinline__ void tdm_load_tile(unsigned lds_byte_off,
                                              const void* lds_dst,
                                              const _Float16* gsrc,
                                              unsigned tensor_d0,   // K extent (elems)
                                              unsigned tensor_d1,   // row extent
                                              unsigned tile_rows,   // BM or BN
                                              unsigned stride0)     // row stride (elems)
{
  unsigned long long ga = (unsigned long long)(uintptr_t)gsrc;
  u32x4 g0;
  g0[0] = 1u;                                            // count=1, user descriptor
  g0[1] = lds_byte_off;                                  // lds_addr
  g0[2] = (unsigned)ga;                                  // global_addr[31:0]
  g0[3] = (unsigned)((ga >> 32) & 0x01FFFFFFu) | (2u << 30); // addr[56:32], type=2
  i32x8 g1;
  g1[0] = (int)((1u << 16)      // data_size = 1 -> 2 bytes
              | (1u << 20)      // pad_enable
              | (3u << 22)      // pad_interval: 16 dwords (64 B)
              | (3u << 25));    // pad_amount:   4 dwords (16 B)
  g1[1] = (int)((tensor_d0 & 0xFFFFu) << 16);                        // dim0 lo
  g1[2] = (int)((tensor_d0 >> 16) | ((tensor_d1 & 0xFFFFu) << 16));  // dim0 hi | dim1 lo
  g1[3] = (int)((tensor_d1 >> 16) | ((unsigned)BK << 16));           // dim1 hi | tile_dim0
  g1[4] = (int)tile_rows;                                            // tile_dim1 (tile_dim2=0)
  g1[5] = (int)stride0;                                              // dim0_stride lo
  g1[6] = 0;
  g1[7] = 0;
  i32x4 z4 = {0, 0, 0, 0};                 // groups 2/3 unused (2-D tensor)
  i32x8 z8 = {0, 0, 0, 0, 0, 0, 0, 0};     // extra group on this toolchain
  __builtin_amdgcn_tensor_load_to_lds(g0, g1, z4, z4, z8, 0);
  asm volatile("" : : "r"(lds_dst) : "memory");   // model the async LDS write
}

// ----------------------- WMMA fragment loads (LDS) -------------------------
// A 16x32 f16 (ISA 7.12.2): lane<16 -> row=lane, K 0..7 (v0-3) & 16..23 (v4-7)
//                           lane>=16 -> row=lane-16, K 8..15 & 24..31
__device__ __forceinline__ v16h load_a_frag(const _Float16* tile, int lane) {
  const int row = lane & 15;
  const int kh  = (lane >> 4) << 3;            // 0 or 8
  const _Float16* p = tile + row * LDK + kh;
  v8h lo = *(const v8h*)(p);
  v8h hi = *(const v8h*)(p + 16);
  return __builtin_shufflevector(lo, hi, 0,1,2,3,4,5,6,7,8,9,10,11,12,13,14,15);
}
// B 32x16 f16: lane<16 -> col=lane, K 0..7 & 16..23; lane>=16 -> col=lane-16, K 8..15 & 24..31
__device__ __forceinline__ v16h load_b_frag(const _Float16* tile, int lane) {
  const int col = lane & 15;
  const int kb  = (lane >> 4) << 3;            // 0 or 8
  const _Float16* p = tile + col * LDK + kb;
  v8h lo = *(const v8h*)(p);
  v8h hi = *(const v8h*)(p + 16);
  return __builtin_shufflevector(lo, hi, 0,1,2,3,4,5,6,7,8,9,10,11,12,13,14,15);
}

// ------------------------------ GEMM kernel --------------------------------
__global__ __launch_bounds__(256) void FusedMoeGemmSilu_wmma(
    const _Float16* __restrict__ xh,   // [E][M][K]
    const _Float16* __restrict__ wt,   // [E][N][K]
    const float*    __restrict__ bias, // [E][N]
    float*          __restrict__ out)  // [E][M][N]
{
  __shared__ _Float16 smem[2 * BM * LDK + 2 * BN * LDK]; // 61,440 B

  const int e    = blockIdx.z;
  const int mblk = blockIdx.y;
  const int nblk = blockIdx.x;
  const int tid  = (int)threadIdx.x;
  const int wave = tid >> 5;
  const int lane = tid & 31;

  const unsigned A_BYTES = (unsigned)(BM * LDK * 2);   // 10240
  const unsigned B_BYTES = (unsigned)(BN * LDK * 2);   // 20480

  const _Float16* Ag = xh + ((size_t)e * M_ + (size_t)mblk * BM) * K_;
  const _Float16* Bg = wt + ((size_t)e * N_ + (size_t)nblk * BN) * K_;

  // Prologue: prefetch tiles 0 and 1 (wave0 -> A, wave1 -> B; EXEC-uniform branch)
  if (wave == 0) {
    tdm_load_tile(0u,      smem,                Ag + 0 * BK, K_, M_, BM, K_);
    tdm_load_tile(A_BYTES, smem + BM * LDK,     Ag + 1 * BK, K_, M_, BM, K_);
  } else if (wave == 1) {
    tdm_load_tile(2u * A_BYTES,           smem + 2 * BM * LDK,            Bg + 0 * BK, K_, N_, BN, K_);
    tdm_load_tile(2u * A_BYTES + B_BYTES, smem + 2 * BM * LDK + BN * LDK, Bg + 1 * BK, K_, N_, BN, K_);
  }

  v8f acc[4][4] = {};   // wave tile 64x64 = 4x4 WMMA accumulators

  const int waveM = wave >> 2;  // 0..1
  const int waveN = wave & 3;   // 0..3

  for (int kt = 0; kt < NKT; ++kt) {
    if (wave < 2) {
      if (kt == NKT - 1) __builtin_amdgcn_s_wait_tensorcnt(0);
      else               __builtin_amdgcn_s_wait_tensorcnt(1); // tile kt done, kt+1 in flight
    }
    __syncthreads();                 // publish tile kt to all waves
    LDS_COMPILER_FENCE();            // force fresh LDS reads this iteration

    const int buf = kt & 1;
    const _Float16* sA = smem + (buf ? BM * LDK : 0);
    const _Float16* sB = smem + 2 * (BM * LDK) + (buf ? BN * LDK : 0);
    const _Float16* aw = sA + (waveM * 64) * LDK;
    const _Float16* bw = sB + (waveN * 64) * LDK;

    v16h af[4], bf[4];
    #pragma unroll
    for (int i = 0; i < 4; ++i) af[i] = load_a_frag(aw + i * 16 * LDK, lane);
    #pragma unroll
    for (int i = 0; i < 4; ++i) bf[i] = load_b_frag(bw + i * 16 * LDK, lane);

    #pragma unroll
    for (int mi = 0; mi < 4; ++mi)
      #pragma unroll
      for (int ni = 0; ni < 4; ++ni)
        acc[mi][ni] = __builtin_amdgcn_wmma_f32_16x16x32_f16(
            false, af[mi], false, bf[ni], (short)0, acc[mi][ni], false, false);

    LDS_COMPILER_FENCE();            // keep LDS reads above the release barrier
    __syncthreads();                 // all waves done reading buf before overwrite
    if (kt + 2 < NKT) {
      if (wave == 0)
        tdm_load_tile(buf ? A_BYTES : 0u,
                      smem + (buf ? BM * LDK : 0),
                      Ag + (size_t)(kt + 2) * BK, K_, M_, BM, K_);
      else if (wave == 1)
        tdm_load_tile(2u * A_BYTES + (buf ? B_BYTES : 0u),
                      smem + 2 * BM * LDK + (buf ? BN * LDK : 0),
                      Bg + (size_t)(kt + 2) * BK, K_, N_, BN, K_);
    }
  }

  // ---------------- Epilogue: bias + SiLU, direct f32 stores ----------------
  // C/D layout: VGPR r -> (M = r | 8+r), N = lane&15 (halves selected by lane>>4)
  const int mBase = mblk * BM + waveM * 64;
  const int nBase = nblk * BN + waveN * 64;
  #pragma unroll
  for (int ni = 0; ni < 4; ++ni) {
    const int col = nBase + ni * 16 + (lane & 15);
    const float bv = bias[(size_t)e * N_ + col];
    #pragma unroll
    for (int mi = 0; mi < 4; ++mi) {
      const int row0 = mBase + mi * 16 + (lane >> 4) * 8;
      float* op = out + ((size_t)e * M_ + row0) * N_ + col;
      v8f a = acc[mi][ni];
      #pragma unroll
      for (int r = 0; r < 8; ++r) {
        float v = a[r] + bv;
        v = v * (1.0f / (1.0f + __expf(-v)));   // silu
        op[(size_t)r * N_] = v;
      }
    }
  }
}

// ------------------------- fp32 -> f16 cast kernels ------------------------
__global__ __launch_bounds__(256) void CvtX_f32_f16(const float* __restrict__ x,
                                                    _Float16* __restrict__ xh) {
  const size_t i = ((size_t)blockIdx.x * 256 + threadIdx.x) * 8;
  f32x4 a = *(const f32x4*)(x + i);
  f32x4 b = *(const f32x4*)(x + i + 4);
  v8h h;
  #pragma unroll
  for (int j = 0; j < 4; ++j) { h[j] = (_Float16)a[j]; h[j + 4] = (_Float16)b[j]; }
  *(v8h*)(xh + i) = h;
}

// w [E][K][N] f32 -> wt [E][N][K] f16, 32x32 LDS tile transpose (coalesced both sides)
__global__ __launch_bounds__(256) void CvtWT_f32_f16(const float* __restrict__ w,
                                                     _Float16* __restrict__ wt) {
  __shared__ _Float16 t[32][33];
  const int e  = blockIdx.z;
  const int kb = blockIdx.y * 32;
  const int nb = blockIdx.x * 32;
  const int tid = (int)threadIdx.x;
  {
    const int kl = tid >> 3, nq = tid & 7;
    f32x4 v = *(const f32x4*)(w + ((size_t)e * K_ + kb + kl) * N_ + nb + nq * 4);
    #pragma unroll
    for (int j = 0; j < 4; ++j) t[nq * 4 + j][kl] = (_Float16)v[j];
  }
  __syncthreads();
  {
    const int nl = tid >> 3, kq = tid & 7;
    v4h h;
    #pragma unroll
    for (int j = 0; j < 4; ++j) h[j] = t[nl][kq * 4 + j];
    *(v4h*)(wt + ((size_t)e * N_ + nb + nl) * K_ + kb + kq * 4) = h;
  }
}

// ------------------------------- launcher ----------------------------------
extern "C" void kernel_launch(void* const* d_in, const int* in_sizes, int n_in,
                              void* d_out, int out_size, void* d_ws, size_t ws_size,
                              hipStream_t stream) {
  (void)in_sizes; (void)n_in; (void)out_size; (void)ws_size;
  const float* x = (const float*)d_in[0];
  const float* w = (const float*)d_in[1];
  const float* b = (const float*)d_in[2];
  float* out = (float*)d_out;

  _Float16* xh = (_Float16*)d_ws;                    // E*M*K f16 = 128 MiB
  _Float16* wt = xh + (size_t)E_ * M_ * K_;          // E*N*K f16 = 256 MiB

  CvtX_f32_f16<<< (unsigned)((size_t)E_ * M_ * K_ / (256 * 8)), 256, 0, stream >>>(x, xh);
  CvtWT_f32_f16<<< dim3(N_ / 32, K_ / 32, E_), 256, 0, stream >>>(w, wt);
  FusedMoeGemmSilu_wmma<<< dim3(N_ / BN, M_ / BM, E_), 256, 0, stream >>>(xh, wt, b, out);
}
